// TransformerTranslatorBlock_66752381714827
// MI455X (gfx1250) — matrix operand order repaired and checked by reading, not compiled, not run on hardware
//
#include <hip/hip_runtime.h>
#include <hip/hip_bf16.h>
#include <stdint.h>

// ---------------- model constants ----------------
#define DM   1024          // d_model
#define NH   16            // heads
#define HD   64            // head dim
#define DFF  4096          // ffn dim
#define BB   2             // batch
#define TT   2048          // seq len
#define MR   (BB*TT)       // 4096 rows

typedef __attribute__((ext_vector_type(16))) _Float16 v16h;
typedef __attribute__((ext_vector_type(8)))  _Float16 v8h;
typedef __attribute__((ext_vector_type(8)))  float    v8f;

// ---------------- WMMA helpers ----------------
// A fragment from pointer already positioned at row*ld + k0 + hi*16.
// Layout: lane r (=M), half hi: a[0..15] = K k0+hi*16 .. +15 (contiguous, matches B)
__device__ inline v16h frag_a(const _Float16* p) {
  v8h lo = *(const v8h*)(p);
  v8h hh = *(const v8h*)(p + 8);
  v16h f;
#pragma unroll
  for (int i = 0; i < 8; ++i) { f[i] = lo[i]; f[i + 8] = hh[i]; }
  return f;
}
// B fragment from pointer already positioned at row*ld + k0 + hi*16 (contiguous 16 halfs).
__device__ inline v16h frag_b(const _Float16* p) {
  return *(const v16h*)(p);
}

// Indexed variants (used by attention).
__device__ inline v16h load_a(const _Float16* __restrict__ base, int ld, int row0, int k0,
                              int r, int hi) {
  return frag_a(base + (size_t)(row0 + r) * ld + k0 + hi * 16);
}
__device__ inline v16h load_b(const _Float16* __restrict__ base, int ld, int n0, int k0,
                              int r, int hi) {
  return frag_b(base + (size_t)(n0 + r) * ld + k0 + hi * 16);
}

__device__ inline v8f wmma16(v16h a, v16h b, v8f c) {
  return __builtin_amdgcn_wmma_f32_16x16x32_f16(false, a, false, b, (short)0, c, false, false);
}

__device__ inline float gelu_tanh(float x) {
  const float k0 = 0.7978845608028654f, k1 = 0.044715f;
  float t = tanhf(k0 * (x + k1 * x * x * x));
  return 0.5f * x * (1.0f + t);
}

// ---------------- weight transpose + cast: W[K][N] f32 -> Wt[N][K] f16 ----------------
__global__ void transpose_cast_kernel(const float* __restrict__ W, _Float16* __restrict__ Wt,
                                      int K, int N) {
  size_t i = (size_t)blockIdx.x * blockDim.x + threadIdx.x;
  size_t tot = (size_t)K * N;
  if (i >= tot) return;
  int n = (int)(i / K);
  int k = (int)(i % K);
  Wt[i] = (_Float16)W[(size_t)k * N + n];
}

// ---------------- RMSNorm rows of f32, emit f16 ----------------
__global__ void rmsnorm_cast_kernel(const float* __restrict__ X, _Float16* __restrict__ Y,
                                    int cols) {
  int row = blockIdx.x;
  const float* xr = X + (size_t)row * cols;
  float ss = 0.f;
  for (int c = threadIdx.x; c < cols; c += blockDim.x) { float v = xr[c]; ss += v * v; }
  __shared__ float red[256];
  red[threadIdx.x] = ss;
  __syncthreads();
  for (int s = 128; s > 0; s >>= 1) {
    if ((int)threadIdx.x < s) red[threadIdx.x] += red[threadIdx.x + s];
    __syncthreads();
  }
  float scale = rsqrtf(red[0] / (float)cols + 1e-6f);
  _Float16* yr = Y + (size_t)row * cols;
  for (int c = threadIdx.x; c < cols; c += blockDim.x) yr[c] = (_Float16)(xr[c] * scale);
}

// ---------------- generic WMMA GEMM: C[M,N] = A[M,K](f16) * Bt[N,K](f16) + bias ----------------
// Workgroup tile 128M x 64N; 8 waves; wave tile 16M x 64N (4 accumulators).
// Branch-free double-buffered K loop (requires K % 64 == 0): the y-set fragments of
// chunk k0+32 load while chunk k0's WMMAs issue, and vice versa. No conditional
// defs, no register rotation -> no spills. __launch_bounds__(.,1) lifts the VGPR cap.
// flags bit0: apply tanh-GELU. resid (f32 [M,N]) optional. outF (f32) / outH (f16) optional.
__global__ __launch_bounds__(256, 1)
void gemm_wmma_kernel(const _Float16* __restrict__ A, const _Float16* __restrict__ Bt,
                      const float* __restrict__ bias, const float* __restrict__ resid,
                      float* __restrict__ outF, _Float16* __restrict__ outH,
                      int M, int N, int K, int flags) {
  int lane = threadIdx.x & 31, wid = threadIdx.x >> 5;
  int r = lane & 15, hi = lane >> 4;
  int m0 = blockIdx.y * 128 + wid * 16;
  int n0 = blockIdx.x * 64;

  const _Float16* Ap = A  + (size_t)(m0 + r) * K + hi * 16;
  const _Float16* B0 = Bt + (size_t)(n0 + r) * K + hi * 16;
  const _Float16* B1 = B0 + (size_t)16 * K;
  const _Float16* B2 = B0 + (size_t)32 * K;
  const _Float16* B3 = B0 + (size_t)48 * K;

  v8f c0 = {}, c1 = {}, c2 = {}, c3 = {};

  // prime x-set with chunk 0
  v16h a0 = frag_a(Ap);
  v16h x0 = frag_b(B0), x1 = frag_b(B1), x2 = frag_b(B2), x3 = frag_b(B3);

  for (int k0 = 0; k0 < K - 64; k0 += 64) {
    // y-set: chunk k0+32 (in flight during x-stage WMMAs)
    v16h a1 = frag_a(Ap + k0 + 32);
    v16h y0 = frag_b(B0 + k0 + 32), y1 = frag_b(B1 + k0 + 32);
    v16h y2 = frag_b(B2 + k0 + 32), y3 = frag_b(B3 + k0 + 32);
    __builtin_prefetch(Ap + k0 + 128, 0, 1);   // global_prefetch_b8, 2 chunks ahead
    __builtin_prefetch(B0 + k0 + 128, 0, 1);
    c0 = wmma16(a0, x0, c0);
    c1 = wmma16(a0, x1, c1);
    c2 = wmma16(a0, x2, c2);
    c3 = wmma16(a0, x3, c3);
    // x-set: chunk k0+64 (in flight during y-stage WMMAs)
    a0 = frag_a(Ap + k0 + 64);
    x0 = frag_b(B0 + k0 + 64); x1 = frag_b(B1 + k0 + 64);
    x2 = frag_b(B2 + k0 + 64); x3 = frag_b(B3 + k0 + 64);
    c0 = wmma16(a1, y0, c0);
    c1 = wmma16(a1, y1, c1);
    c2 = wmma16(a1, y2, c2);
    c3 = wmma16(a1, y3, c3);
  }
  {  // tail: chunks K-64 (x-set, already loaded) and K-32
    v16h a1 = frag_a(Ap + K - 32);
    v16h y0 = frag_b(B0 + K - 32), y1 = frag_b(B1 + K - 32);
    v16h y2 = frag_b(B2 + K - 32), y3 = frag_b(B3 + K - 32);
    c0 = wmma16(a0, x0, c0);
    c1 = wmma16(a0, x1, c1);
    c2 = wmma16(a0, x2, c2);
    c3 = wmma16(a0, x3, c3);
    c0 = wmma16(a1, y0, c0);
    c1 = wmma16(a1, y1, c1);
    c2 = wmma16(a1, y2, c2);
    c3 = wmma16(a1, y3, c3);
  }

#pragma unroll
  for (int v = 0; v < 8; ++v) {
    int row = m0 + v + 8 * hi;
#pragma unroll
    for (int t = 0; t < 4; ++t) {
      int col = n0 + t * 16 + r;
      float acc = (t == 0) ? c0[v] : (t == 1) ? c1[v] : (t == 2) ? c2[v] : c3[v];
      float val = acc + bias[col];
      if (flags & 1) val = gelu_tanh(val);
      size_t idx = (size_t)row * N + col;
      if (resid) val += resid[idx];
      if (outF) outF[idx] = val;
      if (outH) outH[idx] = (_Float16)val;
    }
  }
}

// ---------------- per-head RMSNorm + half-truncated RoPE, split Q/K/V ----------------
// qkv f32 rows [b*T+t][3*DM] ordered (three, head, dim). Writes q/k [b,h,t,d] f16, v transposed [b,h,d,t] f16.
__global__ void qkv_prep_kernel(const float* __restrict__ qkv, _Float16* __restrict__ q16,
                                _Float16* __restrict__ k16, _Float16* __restrict__ vT) {
  int idx = blockIdx.x;            // (b*T + t)*NH + h
  int h  = idx % NH;
  int bt = idx / NH;
  int t  = bt % TT;
  int b  = bt / TT;
  int tid = threadIdx.x;           // 0..63
  const float* row = qkv + (size_t)bt * (3 * DM);
  float qv = row[(0 * NH + h) * HD + tid];
  float kv = row[(1 * NH + h) * HD + tid];
  float vv = row[(2 * NH + h) * HD + tid];

  __shared__ float sq[HD], sk[HD], red[HD];
  // rms(q)
  red[tid] = qv * qv; __syncthreads();
  for (int s = 32; s > 0; s >>= 1) { if (tid < s) red[tid] += red[tid + s]; __syncthreads(); }
  float qs = rsqrtf(red[0] / (float)HD + 1e-6f);
  __syncthreads();
  // rms(k)
  red[tid] = kv * kv; __syncthreads();
  for (int s = 32; s > 0; s >>= 1) { if (tid < s) red[tid] += red[tid + s]; __syncthreads(); }
  float ks = rsqrtf(red[0] / (float)HD + 1e-6f);
  __syncthreads();
  sq[tid] = qv * qs; sk[tid] = kv * ks;
  __syncthreads();

  // RoPE: j in [0,16): freq=(1/1024)^(j/15); j in [16,32): freq=0.
  int j = tid & 31;
  bool upper = tid >= 32;
  float freq = (j < 16) ? powf(1.0f / 1024.0f, (float)j / 15.0f) : 0.0f;
  float th = (float)t * freq;
  float c = __cosf(th), s = __sinf(th);
  float x1q = sq[j], x2q = sq[j + 32];
  float x1k = sk[j], x2k = sk[j + 32];
  float yq = upper ? (-x1q * s + x2q * c) : (x1q * c + x2q * s);
  float yk = upper ? (-x1k * s + x2k * c) : (x1k * c + x2k * s);

  size_t qi = ((size_t)(b * NH + h) * TT + t) * HD + tid;
  q16[qi] = (_Float16)yq;
  k16[qi] = (_Float16)yk;
  vT[((size_t)(b * NH + h) * HD + tid) * TT + t] = (_Float16)vv;
}

// ---------------- flash attention (causal), WMMA-based ----------------
// Q,K: [b,h,t,d] f16 ; Vt: [b,h,d,t] f16 ; Oh: [b,t, h*64+d] f16.
// grid.x = BB*NH*(TT/64); block = 128 (4 waves); each wave owns 16 query rows.
__global__ __launch_bounds__(128, 1)
void attn_kernel(const _Float16* __restrict__ Q, const _Float16* __restrict__ Kh,
                 const _Float16* __restrict__ Vt, _Float16* __restrict__ Oh) {
  int lane = threadIdx.x & 31, wid = threadIdx.x >> 5;
  int r = lane & 15, hi = lane >> 4;
  const int nqb = TT / 64;
  int qb = blockIdx.x % nqb;
  int bh = blockIdx.x / nqb;           // b*NH + h
  int b = bh / NH, h = bh % NH;
  int q0 = qb * 64 + wid * 16;

  const _Float16* Qb = Q  + (size_t)bh * TT * HD;
  const _Float16* Kb = Kh + (size_t)bh * TT * HD;
  const _Float16* Vb = Vt + (size_t)bh * HD * TT;

  __shared__ __attribute__((aligned(32))) _Float16 pst[4][16 * 32];
  _Float16* P = pst[wid];              // per-wave private tile, no cross-wave sync needed

  v16h qa0 = load_a(Qb, HD, q0, 0,  r, hi);
  v16h qa1 = load_a(Qb, HD, q0, 32, r, hi);

  v8f o0 = {}, o1 = {}, o2 = {}, o3 = {};
  float m[8], l[8];
#pragma unroll
  for (int v = 0; v < 8; ++v) { m[v] = -3.0e38f; l[v] = 0.f; }

  int kend = q0 + 16;                  // keys needed: <= q0+15 (causal)
  for (int kb = 0; kb < kend; kb += 32) {
    // S = Q * K^T over head dim (K=64 -> two k32 chunks), two 16-key column tiles
    v8f s0 = {}, s1 = {};
    v16h kb00 = load_b(Kb, HD, kb,      0,  r, hi);
    v16h kb01 = load_b(Kb, HD, kb,      32, r, hi);
    v16h kb10 = load_b(Kb, HD, kb + 16, 0,  r, hi);
    v16h kb11 = load_b(Kb, HD, kb + 16, 32, r, hi);
    // V fragments for this key block, issued early to overlap with S/softmax
    v16h vb0 = load_b(Vb, TT, 0,  kb, r, hi);
    v16h vb1 = load_b(Vb, TT, 16, kb, r, hi);
    v16h vb2 = load_b(Vb, TT, 32, kb, r, hi);
    v16h vb3 = load_b(Vb, TT, 48, kb, r, hi);
    s0 = wmma16(qa0, kb00, s0);
    s0 = wmma16(qa1, kb01, s0);
    s1 = wmma16(qa0, kb10, s1);
    s1 = wmma16(qa1, kb11, s1);

    // online softmax, row-wise across 16 lanes of each half-wave
#pragma unroll
    for (int v = 0; v < 8; ++v) {
      int row = q0 + v + 8 * hi;
      float e0 = s0[v] * 0.125f; if (kb + r      > row) e0 = -3.0e38f;
      float e1 = s1[v] * 0.125f; if (kb + 16 + r > row) e1 = -3.0e38f;
      float mx = fmaxf(e0, e1);
      mx = fmaxf(mx, __shfl_xor(mx, 1, 32));
      mx = fmaxf(mx, __shfl_xor(mx, 2, 32));
      mx = fmaxf(mx, __shfl_xor(mx, 4, 32));
      mx = fmaxf(mx, __shfl_xor(mx, 8, 32));
      float mn = fmaxf(m[v], mx);
      float sc = __expf(m[v] - mn);
      float p0 = __expf(e0 - mn);
      float p1 = __expf(e1 - mn);
      float rs = p0 + p1;
      rs += __shfl_xor(rs, 1, 32);
      rs += __shfl_xor(rs, 2, 32);
      rs += __shfl_xor(rs, 4, 32);
      rs += __shfl_xor(rs, 8, 32);
      l[v] = l[v] * sc + rs;
      m[v] = mn;
      o0[v] = o0[v] * sc; o1[v] = o1[v] * sc; o2[v] = o2[v] * sc; o3[v] = o3[v] * sc;
      // stage P row-major 16x32 in LDS (same-wave DS ops are in order)
      P[(v + 8 * hi) * 32 + r]      = (_Float16)p0;
      P[(v + 8 * hi) * 32 + 16 + r] = (_Float16)p1;
    }

    // O += P(16x32) * V(32x64): reload P as A fragment
    v16h pa = load_a(P, 32, 0, 0, r, hi);
    o0 = wmma16(pa, vb0, o0);
    o1 = wmma16(pa, vb1, o1);
    o2 = wmma16(pa, vb2, o2);
    o3 = wmma16(pa, vb3, o3);
  }

#pragma unroll
  for (int v = 0; v < 8; ++v) {
    int row = q0 + v + 8 * hi;
    float inv = 1.0f / l[v];
    size_t base = ((size_t)b * TT + row) * DM + h * HD;
    Oh[base + 0  + r] = (_Float16)(o0[v] * inv);
    Oh[base + 16 + r] = (_Float16)(o1[v] * inv);
    Oh[base + 32 + r] = (_Float16)(o2[v] * inv);
    Oh[base + 48 + r] = (_Float16)(o3[v] * inv);
  }
}

// ---------------- host orchestration ----------------
extern "C" void kernel_launch(void* const* d_in, const int* in_sizes, int n_in,
                              void* d_out, int out_size, void* d_ws, size_t ws_size,
                              hipStream_t stream) {
  (void)in_sizes; (void)n_in; (void)out_size; (void)ws_size;
  const float* x    = (const float*)d_in[0];
  // d_in[1] = block_mask (causal) — handled analytically in attn_kernel
  const float* Wqkv = (const float*)d_in[2];
  const float* bqkv = (const float*)d_in[3];
  const float* Wo   = (const float*)d_in[4];
  const float* bo   = (const float*)d_in[5];
  const float* W1   = (const float*)d_in[6];
  const float* b1   = (const float*)d_in[7];
  const float* W2   = (const float*)d_in[8];
  const float* b2   = (const float*)d_in[9];
  float* out = (float*)d_out;

  char* p = (char*)d_ws;
  auto carve = [&](size_t n) { char* r = p; p += (n + 255) & ~(size_t)255; return r; };
  _Float16* Wqkv_t = (_Float16*)carve((size_t)3 * DM * DM * 2);
  _Float16* Wo_t   = (_Float16*)carve((size_t)DM * DM * 2);
  _Float16* W1_t   = (_Float16*)carve((size_t)DFF * DM * 2);
  _Float16* W2_t   = (_Float16*)carve((size_t)DM * DFF * 2);
  _Float16* xn16   = (_Float16*)carve((size_t)MR * DM * 2);
  _Float16* q16    = (_Float16*)carve((size_t)MR * DM * 2);
  _Float16* k16    = (_Float16*)carve((size_t)MR * DM * 2);
  _Float16* vT16   = (_Float16*)carve((size_t)MR * DM * 2);
  _Float16* attn16 = (_Float16*)carve((size_t)MR * DM * 2);
  float*    h1     = (float*)carve((size_t)MR * DM * 4);
  char*     scratch = carve((size_t)MR * 3 * DM * 4);   // 48 MB, time-shared:
  float*    qkv32  = (float*)scratch;                   //   phase 1: QKV activations
  _Float16* ff16   = (_Float16*)scratch;                //   phase 2: MLP hidden (32 MB)
  _Float16* h1n16  = (_Float16*)(scratch + (((size_t)MR * DFF * 2 + 255) & ~(size_t)255));

  // 1) weights -> f16 transposed [N][K]
  {
    size_t n;
    n = (size_t)3 * DM * DM;
    transpose_cast_kernel<<<(unsigned)((n + 255) / 256), 256, 0, stream>>>(Wqkv, Wqkv_t, DM, 3 * DM);
    n = (size_t)DM * DM;
    transpose_cast_kernel<<<(unsigned)((n + 255) / 256), 256, 0, stream>>>(Wo, Wo_t, DM, DM);
    n = (size_t)DM * DFF;
    transpose_cast_kernel<<<(unsigned)((n + 255) / 256), 256, 0, stream>>>(W1, W1_t, DM, DFF);
    n = (size_t)DFF * DM;
    transpose_cast_kernel<<<(unsigned)((n + 255) / 256), 256, 0, stream>>>(W2, W2_t, DFF, DM);
  }

  // 2) xn = rms_norm(x) (f16)
  rmsnorm_cast_kernel<<<MR, 256, 0, stream>>>(x, xn16, DM);

  // 3) qkv = xn @ Wqkv + bqkv  (f32)
  gemm_wmma_kernel<<<dim3(3 * DM / 64, MR / 128), 256, 0, stream>>>(
      xn16, Wqkv_t, bqkv, nullptr, qkv32, nullptr, MR, 3 * DM, DM, 0);

  // 4) per-head rmsnorm + rope, split Q/K/V
  qkv_prep_kernel<<<MR * NH, HD, 0, stream>>>(qkv32, q16, k16, vT16);

  // 5) causal flash attention
  attn_kernel<<<BB * NH * (TT / 64), 128, 0, stream>>>(q16, k16, vT16, attn16);

  // 6) h1 = x + attn @ Wo + bo  (f32)
  gemm_wmma_kernel<<<dim3(DM / 64, MR / 128), 256, 0, stream>>>(
      attn16, Wo_t, bo, x, h1, nullptr, MR, DM, DM, 0);

  // 7) h1n = rms_norm(h1) (f16)
  rmsnorm_cast_kernel<<<MR, 256, 0, stream>>>(h1, h1n16, DM);

  // 8) ff = gelu(h1n @ W1 + b1) (f16)
  gemm_wmma_kernel<<<dim3(DFF / 64, MR / 128), 256, 0, stream>>>(
      h1n16, W1_t, b1, nullptr, nullptr, ff16, MR, DFF, DM, 1);

  // 9) out = h1 + ff @ W2 + b2 (f32)
  gemm_wmma_kernel<<<dim3(DM / 64, MR / 128), 256, 0, stream>>>(
      ff16, W2_t, b2, h1, out, nullptr, MR, DM, DFF, 0);
}
